// DecodingSupertagger_35493609735005
// MI455X (gfx1250) — compile-verified, matrix operand order checked by reading, not run on hardware
//
#include <hip/hip_runtime.h>
#include <hip/hip_bf16.h>

#define HH 1024
#define BB 4096
#define AA 128
#define STEPS 50

typedef __attribute__((ext_vector_type(16))) __bf16 v16bf;
typedef __attribute__((ext_vector_type(8)))  float  v8f;

union Frag16 { v16bf v; uint4 q[2]; };
union Acc8   { v8f  v; float f[8]; };

// 32-bit LDS byte offset of a shared-memory pointer (AS3 pointers are 32-bit).
__device__ __forceinline__ uint32_t lds_off(const void* p) {
    return (uint32_t)(uintptr_t)(const __attribute__((address_space(3))) char*)p;
}

// CDNA5 async copy: 16 bytes global -> LDS per lane, tracked by ASYNCcnt.
__device__ __forceinline__ void async_ld16(uint32_t ldsoff, const void* gptr) {
    asm volatile("global_load_async_to_lds_b128 %0, %1, off"
                 :: "v"(ldsoff), "v"(gptr) : "memory");
}
__device__ __forceinline__ void wait_async0() {
    asm volatile("s_wait_asynccnt 0x0" ::: "memory");
}

// Load one bf16 WMMA fragment (16-bit A/B layout): lane<16 holds row=lane,
// K = {0..7, 16..23}; lane>=16 holds row=lane-16, K = {8..15, 24..31}.
// Each half is 8 contiguous bf16 = one ds_load_b128.
__device__ __forceinline__ v16bf load_frag(const __bf16* base, int stride, int lane) {
    const int r  = lane & 15;
    const int kh = (lane >> 4) & 1;
    const __bf16* p = base + r * stride + kh * 8;
    Frag16 f;
    f.q[0] = *(const uint4*)(p);
    f.q[1] = *(const uint4*)(p + 16);
    return f.v;
}

__device__ __forceinline__ float sigm(float x) {
    return 1.0f / (1.0f + __expf(-x));
}

// ---------------------------------------------------------------------------
// Fused gates-GEMM + LSTM cell.  gates = h_in @ W.T + bias  (W is (4H,H) bf16
// row-major).  Block covers 128 rows x 32 gate-cols; computes the tile for all
// 4 gates (col offsets g*1024) so the cell update is local.
// Grid: (H/32, B/128), 256 threads = 8 waves; wave w owns rows w*16..w*16+15.
// K loop manually unrolled x2 so the LDS buffer index is compile-time constant
// (keeps WMMA accumulators register-tied, no v_mov shuffles); B fragments are
// software-pipelined 2 deep so WMMAs wait on dscnt<=2 instead of full drains.
// ---------------------------------------------------------------------------
__global__ __launch_bounds__(256)
void lstm_step_kernel(const __bf16* __restrict__ h_in,   // (B,H) bf16
                      const __bf16* __restrict__ W,      // (4H,H) bf16
                      const float*  __restrict__ bias,   // (4H)
                      float*        __restrict__ c,      // (B,H) f32 in/out
                      __bf16*       __restrict__ h_out)  // (B,H) bf16
{
    constexpr int LS = 40;                       // padded LDS row stride (bf16)
    __shared__ __bf16 sX[2][128 * LS];           // 10 KB per buffer
    __shared__ __bf16 sW[2][4 * 32 * LS];        // 10 KB per buffer

    const int tid  = threadIdx.x;
    const int lane = tid & 31;
    const int wv   = tid >> 5;                   // wave id 0..7
    const int r0   = blockIdx.y * 128;           // row base
    const int j0   = blockIdx.x * 32;            // gate-column base

    const uint32_t sx0 = lds_off(&sX[0][0]);
    const uint32_t sw0 = lds_off(&sW[0][0]);

    Acc8 acc[4][2];
    #pragma unroll
    for (int g = 0; g < 4; ++g)
        #pragma unroll
        for (int nt = 0; nt < 2; ++nt)
            #pragma unroll
            for (int r = 0; r < 8; ++r) acc[g][nt].f[r] = 0.0f;

    // Issue this thread's share of one K-tile: 2 X chunks + 2 W chunks, 16B each.
    auto stage_async = [&](int kt, int buf) {
        const int kc = kt * 32;
        #pragma unroll
        for (int it = 0; it < 2; ++it) {
            const int ch = tid + it * 256;                 // 0..511
            const int xr = ch >> 2, xc = (ch & 3) * 8;
            async_ld16(sx0 + (uint32_t)(buf * 128 * LS + xr * LS + xc) * 2u,
                       h_in + (size_t)(r0 + xr) * HH + kc + xc);
            const int g = ch >> 7, wr = (ch >> 2) & 31, wc = (ch & 3) * 8;
            async_ld16(sw0 + (uint32_t)(buf * 4 * 32 * LS + (g * 32 + wr) * LS + wc) * 2u,
                       W + (size_t)((g << 10) + j0 + wr) * HH + kc + wc);
        }
    };

    // One K-tile of compute from LDS buffer `buf` (call with literal buf!).
    auto compute = [&](int buf) {
        const v16bf afrag = load_frag(&sX[buf][wv * 16 * LS], LS, lane);
        auto ldb = [&](int i) {
            return load_frag(&sW[buf][((i >> 1) * 32 + (i & 1) * 16) * LS], LS, lane);
        };
        v16bf bA = ldb(0);
        v16bf bB = ldb(1);
        #pragma unroll
        for (int i = 0; i < 8; ++i) {
            acc[i >> 1][i & 1].v = __builtin_amdgcn_wmma_f32_16x16x32_bf16(
                false, afrag, false, bA, (short)0, acc[i >> 1][i & 1].v, false, false);
            bA = bB;
            if (i + 2 < 8) bB = ldb(i + 2);
        }
    };

    stage_async(0, 0);
    wait_async0();
    __syncthreads();

    #pragma unroll 1
    for (int kt = 0; kt < 32; kt += 2) {
        stage_async(kt + 1, 1);                  // async into buffer 1
        compute(0);
        wait_async0();
        __syncthreads();

        if (kt + 2 < 32) stage_async(kt + 2, 0); // async into buffer 0
        compute(1);
        wait_async0();
        __syncthreads();
    }

    // Fused LSTM cell epilogue. D layout: lane gives N (lane&15), vgpr r gives
    // M = r + (lane>=16 ? 8 : 0).
    const int mrow = r0 + wv * 16 + ((lane >> 4) ? 8 : 0);
    const int ncol = lane & 15;
    #pragma unroll
    for (int nt = 0; nt < 2; ++nt) {
        const int col = j0 + nt * 16 + ncol;
        const float bi = bias[col];
        const float bf = bias[HH + col];
        const float bg = bias[2 * HH + col];
        const float bo = bias[3 * HH + col];
        #pragma unroll
        for (int r = 0; r < 8; ++r) {
            const size_t idx = (size_t)(mrow + r) * HH + col;
            const float iv = acc[0][nt].f[r] + bi;
            const float fv = acc[1][nt].f[r] + bf;
            const float gv = acc[2][nt].f[r] + bg;
            const float ov = acc[3][nt].f[r] + bo;
            const float cv = sigm(fv) * c[idx] + sigm(iv) * tanhf(gv);
            const float hv = sigm(ov) * tanhf(cv);
            c[idx]     = cv;
            h_out[idx] = (__bf16)hv;
        }
    }
}

// ---------------------------------------------------------------------------
// Prediction GEMM: out = h @ w_pred.T + b_pred   (4096x128, K=1024).
// Grid: B/128 blocks, 256 threads; wave w owns rows w*16, all 128 cols.
// w_pred fragments are identical across blocks -> L2/L0 hot; no LDS needed.
// ---------------------------------------------------------------------------
__global__ __launch_bounds__(256)
void pred_kernel(const __bf16* __restrict__ h,    // (B,H) bf16
                 const __bf16* __restrict__ wp,   // (A,H) bf16
                 const float*  __restrict__ bp,   // (A)
                 float*        __restrict__ out)  // (B,A)
{
    const int tid = threadIdx.x, lane = tid & 31, wv = tid >> 5;
    const int r0  = blockIdx.x * 128;

    Acc8 acc[8];
    #pragma unroll
    for (int nt = 0; nt < 8; ++nt)
        #pragma unroll
        for (int r = 0; r < 8; ++r) acc[nt].f[r] = 0.0f;

    const int arow = r0 + wv * 16 + (lane & 15);
    const int kh   = (lane >> 4) & 1;

    #pragma unroll 1
    for (int kc = 0; kc < HH; kc += 32) {
        Frag16 a;
        const __bf16* pa = h + (size_t)arow * HH + kc + kh * 8;
        a.q[0] = *(const uint4*)(pa);
        a.q[1] = *(const uint4*)(pa + 16);
        #pragma unroll
        for (int nt = 0; nt < 8; ++nt) {
            Frag16 b;
            const __bf16* pb = wp + (size_t)(nt * 16 + (lane & 15)) * HH + kc + kh * 8;
            b.q[0] = *(const uint4*)(pb);
            b.q[1] = *(const uint4*)(pb + 16);
            acc[nt].v = __builtin_amdgcn_wmma_f32_16x16x32_bf16(
                false, a.v, false, b.v, (short)0, acc[nt].v, false, false);
        }
    }

    const int mrow = r0 + wv * 16 + ((lane >> 4) ? 8 : 0);
    #pragma unroll
    for (int nt = 0; nt < 8; ++nt) {
        const int col = nt * 16 + (lane & 15);
        const float bb = bp[col];
        #pragma unroll
        for (int r = 0; r < 8; ++r)
            out[(size_t)(mrow + r) * AA + col] = acc[nt].f[r] + bb;
    }
}

// ---------------------------------------------------------------------------
// One-time conversions: bf16 weights, combined recurrent weight (x==h after
// step 0, so steps 1..49 use Wc = w_ih + w_hh: one GEMM instead of two),
// combined bias, bf16 x0, zeroed cell state.
// ---------------------------------------------------------------------------
__global__ void init_kernel(const float* __restrict__ enc,
                            const float* __restrict__ w_ih,
                            const float* __restrict__ w_hh,
                            const float* __restrict__ b_ih,
                            const float* __restrict__ b_hh,
                            const float* __restrict__ w_pred,
                            __bf16* __restrict__ wih_b,
                            __bf16* __restrict__ wc_b,
                            __bf16* __restrict__ wp_b,
                            float*  __restrict__ bias,
                            __bf16* __restrict__ x0_b,
                            float*  __restrict__ c)
{
    const size_t stride = (size_t)gridDim.x * blockDim.x;
    const size_t total  = (size_t)4 * HH * HH;   // == B*H here too
    for (size_t k = (size_t)blockIdx.x * blockDim.x + threadIdx.x; k < total; k += stride) {
        const float a = w_ih[k];
        wih_b[k] = (__bf16)a;
        wc_b[k]  = (__bf16)(a + w_hh[k]);
        if (k < (size_t)AA * HH) wp_b[k] = (__bf16)w_pred[k];
        if (k < (size_t)4 * HH)  bias[k] = b_ih[k] + b_hh[k];
        x0_b[k] = (__bf16)enc[k];
        c[k]    = 0.0f;
    }
}

extern "C" void kernel_launch(void* const* d_in, const int* in_sizes, int n_in,
                              void* d_out, int out_size, void* d_ws, size_t ws_size,
                              hipStream_t stream) {
    const float* enc    = (const float*)d_in[0];
    const float* w_ih   = (const float*)d_in[1];
    const float* w_hh   = (const float*)d_in[2];
    const float* b_ih   = (const float*)d_in[3];
    const float* b_hh   = (const float*)d_in[4];
    const float* w_pred = (const float*)d_in[5];
    const float* b_pred = (const float*)d_in[6];
    float* out = (float*)d_out;

    char* ws = (char*)d_ws;
    size_t off = 0;
    auto carve = [&](size_t bytes) -> char* {
        char* p = ws + off;
        off += (bytes + 255) & ~(size_t)255;
        return p;
    };
    __bf16* wih_b = (__bf16*)carve((size_t)4 * HH * HH * 2);
    __bf16* wc_b  = (__bf16*)carve((size_t)4 * HH * HH * 2);
    __bf16* wp_b  = (__bf16*)carve((size_t)AA * HH * 2);
    float*  bias  = (float*) carve((size_t)4 * HH * 4);
    __bf16* x0_b  = (__bf16*)carve((size_t)BB * HH * 2);
    __bf16* h0    = (__bf16*)carve((size_t)BB * HH * 2);
    __bf16* h1    = (__bf16*)carve((size_t)BB * HH * 2);
    float*  c     = (float*) carve((size_t)BB * HH * 4);
    (void)ws_size; (void)in_sizes; (void)n_in; (void)out_size;

    init_kernel<<<2048, 256, 0, stream>>>(enc, w_ih, w_hh, b_ih, b_hh, w_pred,
                                          wih_b, wc_b, wp_b, bias, x0_b, c);

    const dim3 sgrid(HH / 32, BB / 128);   // (32, 32)
    for (int t = 0; t < STEPS; ++t) {
        const __bf16* hin = (t == 0) ? x0_b : ((t & 1) ? h0 : h1);
        __bf16*      hout = (t & 1) ? h1 : h0;
        const __bf16* W   = (t == 0) ? wih_b : wc_b;
        lstm_step_kernel<<<sgrid, 256, 0, stream>>>(hin, W, bias, c, hout);
        pred_kernel<<<BB / 128, 256, 0, stream>>>(hout, wp_b, b_pred,
                                                  out + (size_t)t * BB * AA);
    }
}
